// LinkPredict_77790447665348
// MI455X (gfx1250) — matrix operand order repaired and checked by reading, not compiled, not run on hardware
//
#include <hip/hip_runtime.h>

// ---------------- CDNA5 WMMA / TDM types ----------------
typedef __attribute__((ext_vector_type(16))) __bf16 v16bf;
typedef __attribute__((ext_vector_type(8)))  float  v8f;
typedef __attribute__((ext_vector_type(4)))  unsigned int u32x4;
typedef __attribute__((ext_vector_type(8)))  int i32x8;
typedef __attribute__((ext_vector_type(4)))  int i32x4;

#define NN   2048
#define HDIM 256
#define EDG  32768
#define OC   32
#define HC   272   // 256 emb cols + 1 denom col + 15 zero pad -> 17 tiles of 16

static __device__ __forceinline__ unsigned short f32_to_bf16(float f) {
  unsigned int u = __float_as_uint(f);
  unsigned int r = (u + 0x7FFFu + ((u >> 16) & 1u)) >> 16;  // RNE
  return (unsigned short)r;
}

// K striping of the 16-bit A operand (ISA 7.12.2 A 16x32 table).
static __device__ __forceinline__ int kmapA(int v, bool up) {
  int base = (v < 4) ? (2 * v) : (16 + 2 * (v - 4));
  return base + (up ? 8 : 0);
}
// K striping of the 16-bit B operand: same as A (lanes 0-15 K=0..7,16..23; lanes 16-31 K=8..15,24..31).
static __device__ __forceinline__ int kmapB(int v, bool up) {
  int base = (v < 4) ? (2 * v) : (16 + 2 * (v - 4));
  return base + (up ? 8 : 0);
}

// A-tile: row-major, one row per lane (lane&15); (K,K+1) halves = one u32 load.
static __device__ __forceinline__ v16bf load_A_tile(const unsigned short* __restrict__ A,
                                                    int row, int ld, int k0, int lane) {
  union { unsigned int u[8]; v16bf v; } t;
  const bool up = lane >= 16;
  const unsigned int* p = (const unsigned int*)(A + (size_t)row * ld);
#pragma unroll
  for (int v = 0; v < 8; ++v) t.u[v] = p[(k0 + kmapA(v, up)) >> 1];
  return t.v;
}

// B-tile from a TRANSPOSED matrix BT [cols x ldT] in global memory.
static __device__ __forceinline__ v16bf load_B_tileT(const unsigned short* __restrict__ BT,
                                                     int col, int ldT, int k0, int lane) {
  union { unsigned int u[8]; v16bf v; } t;
  const bool up = lane >= 16;
  const unsigned int* p = (const unsigned int*)(BT + (size_t)col * ldT);
#pragma unroll
  for (int v = 0; v < 8; ++v) t.u[v] = p[(k0 + kmapB(v, up)) >> 1];
  return t.v;
}

// B-tile from an LDS panel [cols x 32] (one K-chunk), rows of 32 bf16.
static __device__ __forceinline__ v16bf load_B_lds(const unsigned short* sp,
                                                   int col, int lane) {
  union { unsigned int u[8]; v16bf v; } t;
  const bool up = lane >= 16;
  const unsigned int* p = (const unsigned int*)(sp + col * 32);
#pragma unroll
  for (int v = 0; v < 8; ++v) t.u[v] = p[kmapB(v, up) >> 1];
  return t.v;
}

static __device__ __forceinline__ v8f wmma_bf16(v16bf a, v16bf b, v8f c) {
  return __builtin_amdgcn_wmma_f32_16x16x32_bf16(false, a, false, b, (short)0, c, false, false);
}

// ---- TDM: async-load a [HC x 32] bf16 panel (row stride NN) into LDS ----
// D# per CDNA5 ISA. 2D tile: tile_dim0=32, tile_dim1=HC.
static __device__ __forceinline__ void tdm_load_panel(const unsigned short* gptr,
                                                      unsigned int lds_off) {
  unsigned long long ga = (unsigned long long)(uintptr_t)gptr;
  u32x4 g0;
  g0.x = 1u;                                                     // count=1, no gather
  g0.y = lds_off;                                                // lds_addr (bytes)
  g0.z = (unsigned int)ga;                                       // global_addr[31:0]
  g0.w = (unsigned int)((ga >> 32) & 0x1FFFFFFu) | (2u << 30);   // addr[56:32] | type=2
  i32x8 g1;
  g1[0] = (1 << 16);                                  // data_size=1 (2 bytes)
  g1[1] = (int)((NN & 0xFFFF) << 16);                 // tensor_dim0[15:0] @ 63:48
  g1[2] = (int)((NN >> 16) | ((HC & 0xFFFF) << 16));  // dim0 hi | tensor_dim1 lo
  g1[3] = (int)((HC >> 16) | (32 << 16));             // dim1 hi | tile_dim0 = 32
  g1[4] = (int)HC;                                    // tile_dim1 = HC, tile_dim2 = 0
  g1[5] = (int)NN;                                    // tensor_dim0_stride = NN
  g1[6] = 0;
  g1[7] = 0;
  i32x4 z4 = {0, 0, 0, 0};
  i32x8 z8 = {0, 0, 0, 0, 0, 0, 0, 0};
  __builtin_amdgcn_tensor_load_to_lds(g0, g1, z4, z4, z8, 0);
}

// ------------ Kernel 1a: adj (f32 0/1) -> bf16 0/1, plus transpose ------------
__global__ __launch_bounds__(256) void k_binarize(const float* __restrict__ adj,
                                                  unsigned short* __restrict__ A1,
                                                  unsigned short* __restrict__ A1T) {
  int i = blockIdx.x * 256 + threadIdx.x;
  int r = i >> 11, c = i & (NN - 1);
  unsigned short v = (adj[i] != 0.0f) ? (unsigned short)0x3F80 : (unsigned short)0;
  A1[i] = v;
  A1T[(size_t)c * NN + r] = v;
}

// ---- Kernel 1b: BextT [272 x 2048]: rows 0..255 = emb^T (bf16), row 256 = pos ----
__global__ __launch_bounds__(256) void k_build_bext(const float* __restrict__ emb,
                                                    unsigned short* __restrict__ BextT) {
  __shared__ float red[256];
  int j = blockIdx.x, h = threadIdx.x;
  float v = emb[(size_t)j * HDIM + h];
  red[h] = v;
  __syncthreads();
#pragma unroll
  for (int s = 128; s > 0; s >>= 1) {
    if (h < s) red[h] += red[h + s];
    __syncthreads();
  }
  BextT[(size_t)h * NN + j] = f32_to_bf16(v);
  if (h == 0)           BextT[(size_t)256 * NN + j]       = (red[0] > 0.f) ? (unsigned short)0x3F80 : (unsigned short)0;
  if (h >= 1 && h < 16) BextT[(size_t)(256 + h) * NN + j] = 0;
}

// ------- binarize epilogue helper for one 16x16 accumulator -------
template <bool DO_OR>
static __device__ __forceinline__ void store_bin(v8f acc, unsigned short* __restrict__ Out,
                                                 const unsigned short* __restrict__ orA,
                                                 const unsigned short* __restrict__ orB,
                                                 int mb, int nc, int n) {
#pragma unroll
  for (int r = 0; r < 8; ++r) {
    size_t idx = (size_t)(mb + r) * n + nc;
    bool on = acc[r] > 0.0f;
    if (DO_OR) on = on | (orA[idx] != 0) | (orB[idx] != 0);
    Out[idx] = on ? (unsigned short)0x3F80 : (unsigned short)0;
  }
}

// -------- Kernel 2/3: boolean GEMM, Out = bin(A@B [| orA | orB]) --------
// One wave computes a 16(M) x 64(N) tile: A tile reused across 4 B tiles.
template <bool DO_OR>
__global__ __launch_bounds__(32) void k_gemm_bool(const unsigned short* __restrict__ A,
                                                  const unsigned short* __restrict__ BT,
                                                  const unsigned short* __restrict__ orA,
                                                  const unsigned short* __restrict__ orB,
                                                  unsigned short* __restrict__ Out, int n) {
  const int lane = threadIdx.x & 31;
  const int m0 = blockIdx.x * 16;
  const int n0 = blockIdx.y * 64;
  const int cb = lane & 15;
  const int rowA = m0 + cb;
  const unsigned short* pA = A + (size_t)rowA * n;
  v8f acc0 = {}, acc1 = {}, acc2 = {}, acc3 = {};
  for (int k0 = 0; k0 < n; k0 += 32) {
    if (k0 + 32 < n) __builtin_prefetch(pA + k0 + 32, 0, 1);
    v16bf a = load_A_tile(A, rowA, n, k0, lane);
    acc0 = wmma_bf16(a, load_B_tileT(BT, n0 + cb,      n, k0, lane), acc0);
    acc1 = wmma_bf16(a, load_B_tileT(BT, n0 + 16 + cb, n, k0, lane), acc1);
    acc2 = wmma_bf16(a, load_B_tileT(BT, n0 + 32 + cb, n, k0, lane), acc2);
    acc3 = wmma_bf16(a, load_B_tileT(BT, n0 + 48 + cb, n, k0, lane), acc3);
  }
  const int mb = m0 + ((lane >= 16) ? 8 : 0);
  store_bin<DO_OR>(acc0, Out, orA, orB, mb, n0 + cb,      n);
  store_bin<DO_OR>(acc1, Out, orA, orB, mb, n0 + 16 + cb, n);
  store_bin<DO_OR>(acc2, Out, orA, orB, mb, n0 + 32 + cb, n);
  store_bin<DO_OR>(acc3, Out, orA, orB, mb, n0 + 48 + cb, n);
}

// ---------------- Kernel 4: fused readout-GEMM + conv1d/tanh/fc score ----------
// One block per 16 edges; 8 waves. B panel staged to LDS by TDM (wave 0),
// double-buffered, s_wait_tensorcnt + block barriers for handoff.
__global__ __launch_bounds__(256) void k_score(const unsigned short* __restrict__ Reach,
                                               const unsigned short* __restrict__ BextT,
                                               const float* __restrict__ emb,
                                               const int* __restrict__ src,
                                               const int* __restrict__ rel,
                                               const int* __restrict__ obj,
                                               const float* __restrict__ wrel,
                                               const float* __restrict__ conv_w,
                                               const float* __restrict__ conv_b,
                                               const float* __restrict__ fc_w,
                                               float* __restrict__ out) {
  __shared__ unsigned short s_bt[2][HC * 32];   // 2 x 17408 B TDM panels
  __shared__ float s_sum[16][HC];
  __shared__ float s_part[256];
  __shared__ float s_cw[OC * 4];
  __shared__ float s_cb[OC];

  const int tid = threadIdx.x;
  const int lane = tid & 31;
  const int wave = tid >> 5;                 // 0..7
  const int e0 = blockIdx.x * 16;
  const bool up = lane >= 16;

  if (tid < OC * 4) s_cw[tid] = conv_w[tid];
  if (tid < OC)     s_cb[tid] = conv_b[tid];

  // Reach rows for this lane's edge (packed two bf16 per u32)
  const int er = e0 + (lane & 15);
  const unsigned int* prS = (const unsigned int*)(Reach + (size_t)src[er] * NN);
  const unsigned int* prO = (const unsigned int*)(Reach + (size_t)obj[er] * NN);

  v8f acc0 = {}, acc1 = {}, acc2 = {};
  const int c0 = wave * 16 + (lane & 15);        // tile wave   (<17)
  const int c1 = (wave + 8) * 16 + (lane & 15);  // tile wave+8 (<17)
  const int c2 = 16 * 16 + (lane & 15);          // tile 16 (wave 0 only)

  if (wave == 0)
    tdm_load_panel(BextT, (unsigned int)(uintptr_t)&s_bt[0][0]);

  for (int kk = 0; kk < NN / 32; ++kk) {
    const int cur = kk & 1;
    if (wave == 0) {
      if (kk + 1 < NN / 32) {
        tdm_load_panel(BextT + (kk + 1) * 32,
                       (unsigned int)(uintptr_t)&s_bt[cur ^ 1][0]);
        __builtin_amdgcn_s_wait_tensorcnt(1);   // panel `cur` done, next in flight
      } else {
        __builtin_amdgcn_s_wait_tensorcnt(0);   // last panel done
      }
    }
    __syncthreads();   // panel `cur` ready for all waves

    const int k0 = kk * 32;
    if (k0 + 32 < NN) {
      __builtin_prefetch((const char*)(prS + ((k0 + 32) >> 1)), 0, 1);
      __builtin_prefetch((const char*)(prO + ((k0 + 32) >> 1)), 0, 1);
    }
    union { unsigned int u[8]; v16bf v; } a;
#pragma unroll
    for (int v = 0; v < 8; ++v) {
      int k = k0 + kmapA(v, up);
      a.u[v] = prS[k >> 1] & prO[k >> 1];    // AND of 0/1 bf16 == product
    }
    const unsigned short* sp = &s_bt[cur][0];
    acc0 = wmma_bf16(a.v, load_B_lds(sp, c0, lane), acc0);
    acc1 = wmma_bf16(a.v, load_B_lds(sp, c1, lane), acc1);
    if (wave == 0)
      acc2 = wmma_bf16(a.v, load_B_lds(sp, c2, lane), acc2);

    __syncthreads();   // everyone done reading `cur` before it is overwritten
  }

  // spill accumulators to LDS (C layout: VGPR r -> M = r + 8*(lane>=16), N = lane&15)
  {
    const int mb = up ? 8 : 0;
#pragma unroll
    for (int r = 0; r < 8; ++r) {
      s_sum[mb + r][c0] = acc0[r];
      s_sum[mb + r][c1] = acc1[r];
    }
    if (wave == 0) {
#pragma unroll
      for (int r = 0; r < 8; ++r) s_sum[mb + r][c2] = acc2[r];
    }
  }
  __syncthreads();

  // -------- epilogue: per-edge conv1d(k=1) + tanh + fc dot --------
  const int e = tid >> 4;            // 0..15
  const int hs = (tid & 15) * 16;    // 16 h-values per thread
  float denom = s_sum[e][256];
  if (denom <= 0.f) denom = 1.f;
  const float inv = 1.0f / denom;
  const int eg = e0 + e;
  const float* pS = emb  + (size_t)src[eg] * HDIM;
  const float* pR = wrel + (size_t)rel[eg] * HDIM;
  const float* pO = emb  + (size_t)obj[eg] * HDIM;

  float part = 0.f;
  for (int hh = 0; hh < 16; ++hh) {
    const int h = hs + hh;
    const float sv = pS[h];
    const float rv = pR[h];
    const float ov = pO[h];
    const float ro = s_sum[e][h] * inv;
#pragma unroll
    for (int oc = 0; oc < OC; ++oc) {
      float z = s_cw[oc * 4 + 0] * sv + s_cw[oc * 4 + 1] * rv +
                s_cw[oc * 4 + 2] * ov + s_cw[oc * 4 + 3] * ro + s_cb[oc];
      part += tanhf(z) * fc_w[oc * HDIM + h];
    }
  }
  s_part[tid] = part;
  __syncthreads();
  if (tid < 16) {
    float s = 0.f;
#pragma unroll
    for (int i = 0; i < 16; ++i) s += s_part[tid * 16 + i];
    out[e0 + tid] = -s;
  }
}

// ---------------- host launcher ----------------
extern "C" void kernel_launch(void* const* d_in, const int* in_sizes, int n_in,
                              void* d_out, int out_size, void* d_ws, size_t ws_size,
                              hipStream_t stream) {
  const float* adj    = (const float*)d_in[0];
  const float* emb    = (const float*)d_in[1];
  const int*   src    = (const int*)d_in[2];
  const int*   rel    = (const int*)d_in[3];
  const int*   obj    = (const int*)d_in[4];
  const float* wrel   = (const float*)d_in[5];
  const float* conv_w = (const float*)d_in[6];
  const float* conv_b = (const float*)d_in[7];
  const float* fc_w   = (const float*)d_in[8];
  float* out = (float*)d_out;

  unsigned short* A1    = (unsigned short*)d_ws;                 // 8 MB
  unsigned short* A1T   = A1 + (size_t)NN * NN;                  // 8 MB
  unsigned short* A2    = A1T + (size_t)NN * NN;                 // 8 MB
  unsigned short* Reach = A2 + (size_t)NN * NN;                  // 8 MB
  unsigned short* BextT = Reach + (size_t)NN * NN;               // ~1.1 MB

  k_binarize<<<(NN * NN) / 256, 256, 0, stream>>>(adj, A1, A1T);
  k_build_bext<<<NN, 256, 0, stream>>>(emb, BextT);

  dim3 g(NN / 16, NN / 64);
  // A2 = bin(adj @ adj)            (B columns come from A1T rows)
  k_gemm_bool<false><<<g, 32, 0, stream>>>(A1, A1T, nullptr, nullptr, A2, NN);
  // Reach = bin(A2 @ adj) | A1 | A2   (3-hop reachability)
  k_gemm_bool<true><<<g, 32, 0, stream>>>(A2, A1T, A1, A2, Reach, NN);

  k_score<<<EDG / 16, 256, 0, stream>>>(Reach, BextT, emb, src, rel, obj,
                                        wrel, conv_w, conv_b, fc_w, out);
}
